// train_posterior_49108656062739
// MI455X (gfx1250) — compile-verified, matrix operand order checked by reading, not run on hardware
//
#include <hip/hip_runtime.h>

// ---------------------------------------------------------------------------
// MI455X (gfx1250) fused neg-ELBO kernel, round 2.
// Compute-bound (~65 GFLOP vs ~17MB traffic) -> f16 WMMA (f32 accum).
// Weights are pre-packed into WMMA B-fragment lane layout (1KB per 32x16
// tile, lane l's 16 halves contiguous at l*32 bytes) so each B fragment is
// one coalesced 32B vector load instead of 16 scalar u16 gathers.
// One workgroup (256 thr = 8 wave32) processes 16 samples.
// ---------------------------------------------------------------------------

#define BATCH   32768
#define OBSD    128
#define ADIMN   64
#define ZDIMN   8
#define HDIM    256
#define TS      16          // samples per block
#define NTHREADS 256
#define NWAVES  8

typedef __attribute__((ext_vector_type(16))) _Float16 v16h;
typedef __attribute__((ext_vector_type(8)))  _Float16 v8h;
typedef __attribute__((ext_vector_type(8)))  float    v8f;

// f16 workspace layout (element offsets).  All packed-B regions are arrays of
// 512-element tiles ordered [ntile][ktile].
#define OFF_PW1   0         // 5x16 tiles  (K padded 136->160)
#define OFF_PW2   40960     // 8x16 tiles
#define OFF_PW3   106496    // 8x1 tiles
#define OFF_DW1   110592    // 6x16 tiles
#define OFF_DW2   159744    // 8x16 tiles
#define OFF_DW3   225280    // 8x1 tiles
#define OFF_DW2T  229376    // 8x16 tiles : W2T[k][n] = DW2[n][k]
#define OFF_DW1AT 294912    // 8x4 tiles  : W1aT[k][n] = DW1[128+n][k]
#define OFF_W3MT  311296    // 8x256 row-major (A-side source)
// total 313344 f16 elems = 626688 bytes of d_ws

// ---------------------------------------------------------------------------
// Weight packing: fragment layout per CDNA5 ISA 7.12.2 (16-bit B operand).
// lane l: column n = ntile*16 + (l&15); half = l>>4;
// element e: k = ktile*32 + 8*half + (e&7) + (e>=8 ? 16 : 0)
// ---------------------------------------------------------------------------
__device__ __forceinline__ void pack_b(const float* __restrict__ W, int ld,
                                       int KT, int NT, int kmax, int mode,
                                       _Float16* __restrict__ dst, int g, int stride)
{
    const int total = KT * NT * 512;
    for (int i = g; i < total; i += stride) {
        const int tile = i >> 9, rem = i & 511;
        const int lane = rem >> 4, e = rem & 15;
        const int ntile = tile / KT, ktile = tile - ntile*KT;
        const int half = lane >> 4;
        const int k = ktile*32 + half*8 + (e & 7) + ((e >> 3) << 4);
        const int n = ntile*16 + (lane & 15);
        float v;
        if (k >= kmax)        v = 0.f;
        else if (mode == 0)   v = W[k*ld + n];          // row-major KxN
        else if (mode == 1)   v = W[n*ld + k];          // transpose
        else                  v = W[(128 + n)*ld + k];  // W1a^T slice
        dst[i] = (_Float16)v;
    }
}

__global__ void prep_weights(const float* __restrict__ PW1, const float* __restrict__ PW2,
                             const float* __restrict__ PW3, const float* __restrict__ DW1,
                             const float* __restrict__ DW2, const float* __restrict__ DW3,
                             _Float16* __restrict__ ws)
{
    const int stride = gridDim.x * blockDim.x;
    const int g = blockIdx.x * blockDim.x + threadIdx.x;
    pack_b(PW1, 256, 5, 16, 136, 0, ws + OFF_PW1,   g, stride);
    pack_b(PW2, 256, 8, 16, 256, 0, ws + OFF_PW2,   g, stride);
    pack_b(PW3,  16, 8,  1, 256, 0, ws + OFF_PW3,   g, stride);
    pack_b(DW1, 256, 6, 16, 192, 0, ws + OFF_DW1,   g, stride);
    pack_b(DW2, 256, 8, 16, 256, 0, ws + OFF_DW2,   g, stride);
    pack_b(DW3,  16, 8,  1, 256, 0, ws + OFF_DW3,   g, stride);
    pack_b(DW2, 256, 8, 16, 256, 1, ws + OFF_DW2T,  g, stride);
    pack_b(DW1, 256, 8,  4, 256, 2, ws + OFF_DW1AT, g, stride);
    for (int i = g; i < 8*256; i += stride) {
        const int z = i >> 8, h = i & 255;
        ws[OFF_W3MT + i] = (_Float16)DW3[h*16 + z];
    }
}

// ---------------------------------------------------------------------------
// Fragment loaders
// ---------------------------------------------------------------------------
// Packed B tile: one 32B vector load per lane (coalesced 1KB per wave).
__device__ __forceinline__ v16h load_bp(const _Float16* base, int tile, int lane)
{
    return *(const v16h*)(base + (tile << 9) + (lane << 4));
}

// A fragment from LDS row-major (M=16 rows): two 16B vector loads + shuffle.
__device__ __forceinline__ v16h load_a16(const _Float16* base, int ldk, int k0, int lane)
{
    const int r = lane & 15, half = lane >> 4;
    const _Float16* p = base + r*ldk + k0 + half*8;
    const v8h lo = *(const v8h*)(p);
    const v8h hi = *(const v8h*)(p + 16);
    return __builtin_shufflevector(lo, hi, 0,1,2,3,4,5,6,7,8,9,10,11,12,13,14,15);
}

// Jacobian stage-1 A fragment: rows 0..7 = sample 2p (W3m^T rows masked by
// its D2), rows 8..15 = sample 2p+1.  Vector loads + v_pk_mul_f16 mask.
__device__ __forceinline__ v16h load_t1(const _Float16* W3mT, const _Float16* sD2,
                                        int pair, int k0, int lane)
{
    const int r = lane & 15, half = lane >> 4;
    const int srow = pair*2 + (r >> 3);
    const int z = r & 7;
    const _Float16* w = W3mT + z*HDIM + k0 + half*8;
    const _Float16* m = sD2  + srow*HDIM + k0 + half*8;
    const v8h lo = *(const v8h*)(w)      * *(const v8h*)(m);
    const v8h hi = *(const v8h*)(w + 16) * *(const v8h*)(m + 16);
    return __builtin_shufflevector(lo, hi, 0,1,2,3,4,5,6,7,8,9,10,11,12,13,14,15);
}

__device__ __forceinline__ void store_relu(v8f acc, int n0, const float* bias,
                                           _Float16* sY, _Float16* sMask, int lane)
{
    const int col = n0 + (lane & 15), half = lane >> 4;
    const float bv = bias[col];
#pragma unroll
    for (int i = 0; i < 8; ++i) {
        const int row = i + half*8;          // C layout: M = vgpr_idx + 8*half
        const float pre = acc[i] + bv;
        sY[row*HDIM + col] = (_Float16)fmaxf(pre, 0.f);
        if (sMask) sMask[row*HDIM + col] = (_Float16)(pre > 0.f ? 1.f : 0.f);
    }
}

// One hidden layer: Y(16x256) = relu(X(16xK)@W + b).  Each wave owns two
// N-tiles and shares the A fragment between them (dual accumulators).
template<int KT>
__device__ __forceinline__ void mlp_layer(const _Float16* sX, int ldx,
                                          const _Float16* Wp, const float* bias,
                                          _Float16* sY, _Float16* sMask,
                                          int lane, int wave)
{
    const int t0 = wave*2;
    v8f acc0 = {}, acc1 = {};
#pragma unroll
    for (int kc = 0; kc < KT; ++kc) {
        v16h a  = load_a16(sX, ldx, kc*32, lane);
        v16h b0 = load_bp(Wp, (t0    )*KT + kc, lane);
        v16h b1 = load_bp(Wp, (t0 + 1)*KT + kc, lane);
        acc0 = __builtin_amdgcn_wmma_f32_16x16x32_f16(false, a, false, b0,
                                                      (short)0, acc0, false, false);
        acc1 = __builtin_amdgcn_wmma_f32_16x16x32_f16(false, a, false, b1,
                                                      (short)0, acc1, false, false);
    }
    store_relu(acc0, t0*16,      bias, sY, sMask, lane);
    store_relu(acc1, t0*16 + 16, bias, sY, sMask, lane);
}

// Head layer: O(16x16 f32) = X(16x256)@W(256x16) + b ; wave 0 only.
__device__ __forceinline__ void mlp_head(const _Float16* sX, const _Float16* Wp,
                                         const float* bias, float* sO,
                                         int lane, int wave)
{
    if (wave != 0) return;
    v8f acc = {};
#pragma unroll
    for (int kc = 0; kc < 8; ++kc) {
        v16h a = load_a16(sX, HDIM, kc*32, lane);
        v16h b = load_bp(Wp, kc, lane);
        acc = __builtin_amdgcn_wmma_f32_16x16x32_f16(false, a, false, b,
                                                     (short)0, acc, false, false);
    }
    const int col = lane & 15, half = lane >> 4;
    const float bv = bias[col];
#pragma unroll
    for (int i = 0; i < 8; ++i) sO[(i + half*8)*16 + col] = acc[i] + bv;
}

__device__ __forceinline__ float softplusf_(float x)
{
    return fmaxf(x, 0.f) + log1pf(expf(-fabsf(x)));
}

// ---------------------------------------------------------------------------
// Main fused kernel: 16 samples / block
// ---------------------------------------------------------------------------
__global__ __launch_bounds__(NTHREADS)
void fused_elbo_kernel(const float* __restrict__ obs,   const float* __restrict__ yprime,
                       const float* __restrict__ eps,   const float* __restrict__ mean_action,
                       const float* __restrict__ std_action, const float* __restrict__ power_param,
                       const float* __restrict__ Pb1, const float* __restrict__ Pb2,
                       const float* __restrict__ Pb3, const float* __restrict__ Db1,
                       const float* __restrict__ Db2, const float* __restrict__ Db3,
                       const _Float16* __restrict__ ws, float* __restrict__ out)
{
    __shared__ __align__(32) _Float16 sXin [TS*192];   // input (posterior K<160)
    __shared__ __align__(32) _Float16 sBufA[TS*HDIM];  // h1
    __shared__ __align__(32) _Float16 sBufB[TS*HDIM];  // h2
    __shared__ __align__(32) _Float16 sD1  [TS*HDIM];  // relu mask layer1 (dyn @0)
    __shared__ __align__(32) _Float16 sD2  [TS*HDIM];  // relu mask layer2
    __shared__ __align__(32) _Float16 sT2  [NWAVES*16*32]; // per-wave C->A staging
    __shared__ float    sJac [TS*ZDIMN*ADIMN];         // 16 x 8 x 64
    __shared__ float    sP3  [TS*16];                  // head outputs
    __shared__ float    sZm[TS*ZDIMN], sZlv[TS*ZDIMN], sZ[TS*ZDIMN];
    __shared__ float    sAct [TS*ADIMN];
    __shared__ float    sA0[ADIMN], sCvec[ADIMN], sMean[ADIMN], sStd[ADIMN];

    const int tid  = threadIdx.x;
    const int lane = tid & 31;
    const int wave = tid >> 5;
    const int s0   = blockIdx.x * TS;

    if (tid == 0) {                                    // warm L2 path for weights
        __builtin_prefetch(ws + OFF_DW2T, 0, 1);
        __builtin_prefetch(ws + OFF_PW2, 0, 1);
    }

    // per-action-dim constants: a(0) and d a/d u |_{u=0}
    if (tid < ADIMN) {
        const float m = mean_action[tid], sd = std_action[tid];
        const float sig0 = 1.f / (1.f + expf(2.5f));   // sigmoid(-2.5)
        sMean[tid] = m;  sStd[tid] = sd;
        sA0[tid]   = (sig0 - m) / sd;
        sCvec[tid] = 5.f * sig0 * (1.f - sig0) / sd;
    }
    // posterior input: [obs | y'] zero-padded (ld = 192, K reads stop at 160)
    for (int i = tid; i < TS*192; i += NTHREADS) {
        const int s = i / 192, c = i - s*192;
        float v = (c < OBSD) ? obs[(s0+s)*OBSD + c]
                : (c < OBSD + ZDIMN) ? yprime[(s0+s)*ZDIMN + (c - OBSD)] : 0.f;
        sXin[i] = (_Float16)v;
    }
    __syncthreads();

    // ---- posterior MLP ----
    mlp_layer<5>(sXin, 192, ws + OFF_PW1, Pb1, sBufA, nullptr, lane, wave);
    __syncthreads();
    mlp_layer<8>(sBufA, HDIM, ws + OFF_PW2, Pb2, sBufB, nullptr, lane, wave);
    __syncthreads();
    mlp_head(sBufB, ws + OFF_PW3, Pb3, sP3, lane, wave);
    __syncthreads();

    // z = zm + exp(0.5*zlv)*eps
    if (tid < TS*ZDIMN) {
        const int s = tid >> 3, i = tid & 7;
        const float zm  = sP3[s*16 + i];
        const float zsv = sP3[s*16 + 8 + i];
        const float zlv = 2.f * logf(softplusf_(zsv) + 1e-6f);
        sZm[tid] = zm;  sZlv[tid] = zlv;
        sZ[tid]  = zm + expf(0.5f * zlv) * eps[(s0+s)*ZDIMN + i];
    }
    // dynamics input at u = 0
    for (int i = tid; i < TS*192; i += NTHREADS) {
        const int s = i / 192, c = i - s*192;
        const float v = (c < OBSD) ? obs[(s0+s)*OBSD + c] : sA0[c - OBSD];
        sXin[i] = (_Float16)v;
    }
    __syncthreads();

    // ---- dynamics forward @ u=0 (records relu masks) ----
    mlp_layer<6>(sXin, 192, ws + OFF_DW1, Db1, sBufA, sD1, lane, wave);
    __syncthreads();
    mlp_layer<8>(sBufA, HDIM, ws + OFF_DW2, Db2, sBufB, sD2, lane, wave);
    __syncthreads();

    // ---- Jacobian: J = ((W3m^T ⊙ D2) · W2^T) ⊙ D1 · W1a^T · diag(c)
    //      one wave per sample pair, fully on WMMA ----
    {
        const int pair = wave;
        _Float16* myT2 = sT2 + wave*16*32;
        v8f jacc[4] = {};
        for (int kc = 0; kc < 8; ++kc) {          // K-chunk over h1 (stage-2 K)
            v8f acc0 = {}, acc1 = {};
#pragma unroll
            for (int kk = 0; kk < 8; ++kk) {      // K over h2
                v16h a  = load_t1(ws + OFF_W3MT, sD2, pair, kk*32, lane);
                v16h b0 = load_bp(ws + OFF_DW2T, (kc*2    )*8 + kk, lane);
                v16h b1 = load_bp(ws + OFF_DW2T, (kc*2 + 1)*8 + kk, lane);
                acc0 = __builtin_amdgcn_wmma_f32_16x16x32_f16(false, a, false, b0,
                                                              (short)0, acc0, false, false);
                acc1 = __builtin_amdgcn_wmma_f32_16x16x32_f16(false, a, false, b1,
                                                              (short)0, acc1, false, false);
            }
            const int nl = lane & 15, half = lane >> 4;
#pragma unroll
            for (int i = 0; i < 8; ++i) {         // mask by D1, stage C->A in LDS
                const int row  = i + half*8;
                const int srow = pair*2 + (row >> 3);
                const float m0f = (float)sD1[srow*HDIM + kc*32 + nl];
                const float m1f = (float)sD1[srow*HDIM + kc*32 + 16 + nl];
                myT2[row*32 + nl]      = (_Float16)(acc0[i] * m0f);
                myT2[row*32 + 16 + nl] = (_Float16)(acc1[i] * m1f);
            }
            // same-wave LDS in-order: re-read tile as A fragment
            v16h a2 = load_a16(myT2, 32, 0, lane);
#pragma unroll
            for (int nt = 0; nt < 4; ++nt) {
                v16h b2 = load_bp(ws + OFF_DW1AT, nt*8 + kc, lane);
                jacc[nt] = __builtin_amdgcn_wmma_f32_16x16x32_f16(false, a2, false, b2,
                                                                  (short)0, jacc[nt], false, false);
            }
        }
        const int half = lane >> 4;
#pragma unroll
        for (int nt = 0; nt < 4; ++nt) {
#pragma unroll
            for (int i = 0; i < 8; ++i) {
                const int row  = i + half*8;
                const int srow = pair*2 + (row >> 3);
                const int z    = row & 7;
                const int col  = nt*16 + (lane & 15);
                sJac[(srow*ZDIMN + z)*ADIMN + col] = jacc[nt][i] * sCvec[col];
            }
        }
    }
    __syncthreads();

    // ---- per-sample: eig(J J^T) -> right singular vectors -> action ----
    if (tid < TS) {
        const int s = tid;
        const float* J = &sJac[s*ZDIMN*ADIMN];
        float G[8][8], V[8][8];
        for (int i = 0; i < 8; ++i)
            for (int j = i; j < 8; ++j) {
                float t = 0.f;
                for (int k = 0; k < ADIMN; ++k) t += J[i*ADIMN + k] * J[j*ADIMN + k];
                G[i][j] = t; G[j][i] = t;
            }
        for (int i = 0; i < 8; ++i)
            for (int j = 0; j < 8; ++j) V[i][j] = (i == j) ? 1.f : 0.f;
        for (int sweep = 0; sweep < 8; ++sweep) {
            for (int p = 0; p < 7; ++p)
                for (int q = p + 1; q < 8; ++q) {
                    const float apq = G[p][q];
                    if (fabsf(apq) <= 1e-18f) continue;
                    const float tau = (G[q][q] - G[p][p]) / (2.f * apq);
                    const float t = (tau >= 0.f ? 1.f : -1.f) /
                                    (fabsf(tau) + sqrtf(1.f + tau*tau));
                    const float c = 1.f / sqrtf(1.f + t*t);
                    const float sn = t * c;
                    for (int k = 0; k < 8; ++k) {
                        const float gpk = G[p][k], gqk = G[q][k];
                        G[p][k] = c*gpk - sn*gqk;  G[q][k] = sn*gpk + c*gqk;
                    }
                    for (int k = 0; k < 8; ++k) {
                        const float gkp = G[k][p], gkq = G[k][q];
                        G[k][p] = c*gkp - sn*gkq;  G[k][q] = sn*gkp + c*gkq;
                    }
                    for (int k = 0; k < 8; ++k) {
                        const float vkp = V[k][p], vkq = V[k][q];
                        V[k][p] = c*vkp - sn*vkq;  V[k][q] = sn*vkp + c*vkq;
                    }
                }
        }
        int ord[8];
        for (int i = 0; i < 8; ++i) ord[i] = i;
        for (int i = 0; i < 7; ++i) {             // sort eigenvalues descending
            int best = i;
            for (int j = i + 1; j < 8; ++j)
                if (G[ord[j]][ord[j]] > G[ord[best]][ord[best]]) best = j;
            const int tswap = ord[i]; ord[i] = ord[best]; ord[best] = tswap;
        }
        for (int k = 0; k < ADIMN; ++k) sAct[s*ADIMN + k] = 0.f;
        for (int r = 0; r < 8; ++r) {
            const int e = ord[r];
            const float sv  = sqrtf(fmaxf(G[e][e], 0.f));
            const float inv = (sv > 1e-20f) ? 1.f / sv : 0.f;
            float d = 0.f;                        // diag(J Vh^T)_r for sign fix
            for (int k = 0; k < ADIMN; ++k) {
                float t = 0.f;
                for (int m = 0; m < 8; ++m) t += V[m][e] * J[m*ADIMN + k];
                d += t * J[r*ADIMN + k];
            }
            const float coef = ((d < 0.f) ? -1.f : 1.f) * inv * sZ[s*ZDIMN + r];
            for (int k = 0; k < ADIMN; ++k) {
                float t = 0.f;
                for (int m = 0; m < 8; ++m) t += V[m][e] * J[m*ADIMN + k];
                sAct[s*ADIMN + k] += coef * t;
            }
        }
    }
    __syncthreads();

    // dynamics input with squashed action
    for (int i = tid; i < TS*192; i += NTHREADS) {
        const int s = i / 192, c = i - s*192;
        float v;
        if (c < OBSD) v = obs[(s0+s)*OBSD + c];
        else {
            const int j = c - OBSD;
            const float u = sAct[s*ADIMN + j];
            const float sg = 1.f / (1.f + expf(-(u - 0.5f) * 5.f));
            v = (sg - sMean[j]) / sStd[j];
        }
        sXin[i] = (_Float16)v;
    }
    __syncthreads();

    // ---- dynamics forward @ action ----
    mlp_layer<6>(sXin, 192, ws + OFF_DW1, Db1, sBufA, nullptr, lane, wave);
    __syncthreads();
    mlp_layer<8>(sBufA, HDIM, ws + OFF_DW2, Db2, sBufB, nullptr, lane, wave);
    __syncthreads();
    mlp_head(sBufB, ws + OFF_DW3, Db3, sP3, lane, wave);
    __syncthreads();

    // ---- final reductions ----
    if (tid < TS) {
        const int s = tid;
        const float plv = 2.f * logf(softplusf_(power_param[0]) + 1e-6f);
        const float LOG2PI = 1.8378770664093453f;
        float ll = 0.f, kl = 0.f;
        for (int i = 0; i < ZDIMN; ++i) {
            const float dm  = sP3[s*16 + i];
            const float dlv = sP3[s*16 + 8 + i];
            const float resid = yprime[(s0+s)*ZDIMN + i] - obs[(s0+s)*OBSD + i] - dm;
            ll += -0.5f * (dlv + LOG2PI + resid*resid*expf(-dlv));
            const float zlv = sZlv[s*ZDIMN + i], zm = sZm[s*ZDIMN + i];
            kl += 0.5f * (plv - zlv + expf(zlv - plv) - 1.f + zm*zm*expf(-plv));
        }
        ll *= (1.f / ZDIMN);  kl *= (1.f / ZDIMN);
        out[s0 + s]         = -(ll - kl);
        out[BATCH + s0 + s] = kl;
    }
}

// ---------------------------------------------------------------------------
extern "C" void kernel_launch(void* const* d_in, const int* in_sizes, int n_in,
                              void* d_out, int out_size, void* d_ws, size_t ws_size,
                              hipStream_t stream)
{
    (void)in_sizes; (void)n_in; (void)out_size; (void)ws_size; // ws needs >= 626688 B

    const float* obs = (const float*)d_in[0];
    const float* yp  = (const float*)d_in[1];
    const float* eps = (const float*)d_in[2];
    const float* ma  = (const float*)d_in[3];
    const float* sa  = (const float*)d_in[4];
    const float* pp  = (const float*)d_in[5];
    const float* PW1 = (const float*)d_in[6];
    const float* Pb1 = (const float*)d_in[7];
    const float* PW2 = (const float*)d_in[8];
    const float* Pb2 = (const float*)d_in[9];
    const float* PW3 = (const float*)d_in[10];
    const float* Pb3 = (const float*)d_in[11];
    const float* DW1 = (const float*)d_in[12];
    const float* Db1 = (const float*)d_in[13];
    const float* DW2 = (const float*)d_in[14];
    const float* Db2 = (const float*)d_in[15];
    const float* DW3 = (const float*)d_in[16];
    const float* Db3 = (const float*)d_in[17];
    _Float16* ws = (_Float16*)d_ws;
    float* out = (float*)d_out;

    prep_weights<<<128, 256, 0, stream>>>(PW1, PW2, PW3, DW1, DW2, DW3, ws);
    fused_elbo_kernel<<<BATCH/TS, NTHREADS, 0, stream>>>(
        obs, yp, eps, ma, sa, pp, Pb1, Pb2, Pb3, Db1, Db2, Db3, ws, out);
}
